// RUMC_72980084294363
// MI455X (gfx1250) — matrix operand order repaired and checked by reading, not compiled, not run on hardware
//
#include <hip/hip_runtime.h>
#include <hip/hip_bf16.h>

typedef __attribute__((ext_vector_type(16))) _Float16 v16h;
typedef __attribute__((ext_vector_type(8)))  float    v8f;

// Problem constants (fixed by the reference)
constexpr int   SEQ   = 5;
constexpr int   EMB   = 64;
constexpr int   CLU   = 64;
constexpr float ALPHA = 0.2f;
constexpr float BETA  = 1.0f;
constexpr float KDEC  = 0.1f;

// LDS padding strides (avoid 16-way bank conflicts on 128B-stride rows)
constexpr int WSTR = 66;   // halfs per W row
constexpr int ASTR = 66;   // halfs per aeY row
constexpr int VSTR = 17;   // floats per V row
constexpr int WAVES = 4;
constexpr int TILE_B = 16; // batch elements per wave

__global__ __launch_bounds__(128) void rumc_kernel(
    const int*  __restrict__ u,
    const int*  __restrict__ X,
    const int*  __restrict__ y,
    const float* __restrict__ item_emb,
    const float* __restrict__ user_emb,
    const float* __restrict__ ae_raw,
    const float* __restrict__ w_raw,
    float* __restrict__ out)
{
    __shared__ _Float16 sW[CLU * WSTR];              // softmaxed W (f16), row-major [i][j]
    __shared__ _Float16 sAeY[WAVES][TILE_B * ASTR];  // ae_y per tile,  [b][j]
    __shared__ float    sV[WAVES][CLU * VSTR];       // V = W @ ae_y,   [j][b]

    const int tid  = threadIdx.x;
    const int wave = tid >> 5;
    const int lane = tid & 31;

    // ---- Step 0: W = softmax(500 * w_raw, axis=0) -> f16 in LDS (cooperative) ----
    if (tid < CLU) {
        const int j = tid;
        float m = -1e30f;
        for (int i = 0; i < CLU; ++i) m = fmaxf(m, w_raw[i * CLU + j]);
        float S = 0.f;
        for (int i = 0; i < CLU; ++i) {
            float e = __expf((w_raw[i * CLU + j] - m) * 500.f);
            S += e;
            sW[i * WSTR + j] = (_Float16)e;
        }
        const float inv = 1.f / S;
        for (int i = 0; i < CLU; ++i)
            sW[i * WSTR + j] = (_Float16)((float)sW[i * WSTR + j] * inv);
    }
    __syncthreads();

    // Wave handles 16 batch elements; lane pair (2b,2b+1) owns batch b,
    // each lane owns 32 of the 64 channels.
    const int bbase = blockIdx.x * (WAVES * TILE_B) + wave * TILE_B;
    const int b    = lane >> 1;
    const int half = lane & 1;
    const int gb   = bbase + b;

    // ---- Step 1: aeY[b][:] = softmax(50 * ae_raw[y[b]]) (f16, per-wave LDS) ----
    {
        const int yi = y[gb];
        const float4* p4 = (const float4*)(ae_raw + (size_t)yi * CLU + half * 32);
        float r[32];
        #pragma unroll
        for (int q = 0; q < 8; ++q) {
            float4 v4 = p4[q];
            r[4*q+0] = v4.x; r[4*q+1] = v4.y; r[4*q+2] = v4.z; r[4*q+3] = v4.w;
        }
        float m = -1e30f;
        #pragma unroll
        for (int k = 0; k < 32; ++k) m = fmaxf(m, r[k]);
        m = fmaxf(m, __shfl_xor(m, 1, 32));
        float S = 0.f;
        #pragma unroll
        for (int k = 0; k < 32; ++k) { r[k] = __expf((r[k] - m) * 50.f); S += r[k]; }
        S += __shfl_xor(S, 1, 32);
        const float inv = 1.f / S;
        _Float16* dst = &sAeY[wave][b * ASTR + half * 32];
        #pragma unroll
        for (int k = 0; k < 32; ++k) dst[k] = (_Float16)(r[k] * inv);
    }
    // LDS ops are in-order within a wave; sAeY/sV are private per wave.

    // ---- Step 2: V[64x16] = W[64x64] @ AeY[64x16] via v_wmma_f32_16x16x32_f16 ----
    {
        const int n  = lane & 15;
        const int hi = lane >> 4;

        v16h Bf0, Bf1; // B fragments for k-steps 0 and 1 (shared across M tiles)
        #pragma unroll
        for (int p = 0; p < 8; ++p) {
            const int kk0 = ((p < 4) ? 2 * p : 16 + 2 * (p - 4)) + hi * 8;       // K for kt=0
            const int kk1 = kk0 + 32;              // K for kt=1
            Bf0[2*p]   = sAeY[wave][n * ASTR + kk0];
            Bf0[2*p+1] = sAeY[wave][n * ASTR + kk0 + 1];
            Bf1[2*p]   = sAeY[wave][n * ASTR + kk1];
            Bf1[2*p+1] = sAeY[wave][n * ASTR + kk1 + 1];
        }

        #pragma unroll
        for (int mt = 0; mt < 4; ++mt) {
            const int m = mt * 16 + n;  // A-fragment: lane = M row
            v8f acc = {};
            #pragma unroll
            for (int kt = 0; kt < 2; ++kt) {
                v16h Af;
                #pragma unroll
                for (int p = 0; p < 8; ++p) {
                    // 16-bit A layout: VGPR p holds K pair; lanes 16-31 get K+8
                    const int kk = ((p < 4) ? 2 * p : 16 + 2 * (p - 4)) + hi * 8 + kt * 32;
                    Af[2*p]   = sW[m * WSTR + kk];
                    Af[2*p+1] = sW[m * WSTR + kk + 1];
                }
                acc = __builtin_amdgcn_wmma_f32_16x16x32_f16(
                        false, Af, false, (kt == 0 ? Bf0 : Bf1),
                        (short)0, acc, false, false);
            }
            #pragma unroll
            for (int r = 0; r < 8; ++r)
                sV[wave][(mt * 16 + r + 8 * hi) * VSTR + n] = acc[r];
        }
    }

    // ---- Step 3: per-batch dots, softmaxes, output (lane pairs) ----
    {
        const int yi = y[gb];
        const int ui = u[gb];

        float yv[32];
        if (yi == 0) {
            #pragma unroll
            for (int k = 0; k < 32; ++k) yv[k] = 0.f;   // padding_idx = 0
        } else {
            const float4* p4 = (const float4*)(item_emb + (size_t)yi * EMB + half * 32);
            #pragma unroll
            for (int q = 0; q < 8; ++q) {
                float4 v4 = p4[q];
                yv[4*q+0] = v4.x; yv[4*q+1] = v4.y; yv[4*q+2] = v4.z; yv[4*q+3] = v4.w;
            }
        }

        // u_emb . y_emb
        float uacc = 0.f;
        {
            const float4* p4 = (const float4*)(user_emb + (size_t)ui * EMB + half * 32);
            #pragma unroll
            for (int q = 0; q < 8; ++q) {
                float4 v4 = p4[q];
                uacc += v4.x * yv[4*q+0] + v4.y * yv[4*q+1]
                      + v4.z * yv[4*q+2] + v4.w * yv[4*q+3];
            }
        }
        uacc += __shfl_xor(uacc, 1, 32);

        float fac[SEQ], s1[SEQ];
        #pragma unroll
        for (int s = 0; s < SEQ; ++s) {
            const int xi = X[gb * SEQ + s];

            // fac[s] = softmax(50*ae_raw[xi]) . V[:,b]  (unnormalized-exp trick)
            {
                const float4* p4 = (const float4*)(ae_raw + (size_t)xi * CLU + half * 32);
                float r[32];
                #pragma unroll
                for (int q = 0; q < 8; ++q) {
                    float4 v4 = p4[q];
                    r[4*q+0] = v4.x; r[4*q+1] = v4.y; r[4*q+2] = v4.z; r[4*q+3] = v4.w;
                }
                float m = -1e30f;
                #pragma unroll
                for (int k = 0; k < 32; ++k) m = fmaxf(m, r[k]);
                m = fmaxf(m, __shfl_xor(m, 1, 32));
                float d = 0.f, S = 0.f;
                #pragma unroll
                for (int k = 0; k < 32; ++k) {
                    float e = __expf((r[k] - m) * 50.f);
                    d += e * sV[wave][(half * 32 + k) * VSTR + b];
                    S += e;
                }
                d += __shfl_xor(d, 1, 32);
                S += __shfl_xor(S, 1, 32);
                fac[s] = d / S;
            }

            // s1[s] = X_emb[b,s] . y_emb[b]
            float xd = 0.f;
            if (xi != 0) {   // pair-uniform branch
                const float4* p4 = (const float4*)(item_emb + (size_t)xi * EMB + half * 32);
                #pragma unroll
                for (int q = 0; q < 8; ++q) {
                    float4 v4 = p4[q];
                    xd += v4.x * yv[4*q+0] + v4.y * yv[4*q+1]
                        + v4.z * yv[4*q+2] + v4.w * yv[4*q+3];
                }
            }
            xd += __shfl_xor(xd, 1, 32);
            s1[s] = xd;
        }

        // scores -> softmax over SEQ; p.y == sum_s z_s * s1_s
        float sc[SEQ], msc = -1e30f;
        #pragma unroll
        for (int s = 0; s < SEQ; ++s) {
            const float tf = 1.0f - (float)(SEQ - 1 - s) * KDEC;
            sc[s] = s1[s] * BETA * __expf(fac[s]) * tf;
            msc = fmaxf(msc, sc[s]);
        }
        float Z = 0.f, dotpy = 0.f;
        #pragma unroll
        for (int s = 0; s < SEQ; ++s) {
            sc[s] = __expf(sc[s] - msc);
            Z += sc[s];
            dotpy += sc[s] * s1[s];
        }
        dotpy /= Z;

        const float val = ALPHA * dotpy + uacc;
        const float o = 1.f / (1.f + __expf(-val));
        if (half == 0) out[gb] = o;
    }
}

extern "C" void kernel_launch(void* const* d_in, const int* in_sizes, int n_in,
                              void* d_out, int out_size, void* d_ws, size_t ws_size,
                              hipStream_t stream) {
    const int*   u        = (const int*)  d_in[0];
    const int*   X        = (const int*)  d_in[1];
    const int*   y        = (const int*)  d_in[2];
    const float* item_emb = (const float*)d_in[3];
    const float* user_emb = (const float*)d_in[4];
    const float* ae_raw   = (const float*)d_in[5];
    const float* w_raw    = (const float*)d_in[6];
    float*       out      = (float*)      d_out;

    const int B = in_sizes[0];            // 16384; divisible by 64
    const int blocks = B / (WAVES * TILE_B);
    rumc_kernel<<<blocks, 128, 0, stream>>>(u, X, y, item_emb, user_emb,
                                            ae_raw, w_raw, out);
}
